// ClusterlingLayer_7103875908094
// MI455X (gfx1250) — matrix operand order repaired and checked by reading, not compiled, not run on hardware
//
#include <hip/hip_runtime.h>

#define B_ROWS 8192
#define D_DIM  512
#define K_CL   1024

#define BM 128
#define BN 128
#define BK 32
#define NCHUNK (D_DIM / BK)      // 16
#define TSTRIDE 80               // 64 data bytes per tile row + 16 pad (bank-conflict-free)

typedef __attribute__((ext_vector_type(16))) __bf16 v16bf;
typedef __attribute__((ext_vector_type(8)))  float  v8f;
typedef int v4i_t __attribute__((__vector_size__(16)));

__device__ __forceinline__ unsigned short f32_to_bf16_rne(float f) {
  union { float f; unsigned int u; } c; c.f = f;
  unsigned int u = c.u;
  unsigned int r = u + 0x7FFFu + ((u >> 16) & 1u);
  if ((u & 0x7F800000u) == 0x7F800000u) r = u;   // inf/nan passthrough
  return (unsigned short)(r >> 16);
}

// 16-byte global -> LDS copy. Prefer CDNA5 async path (ASYNCcnt-tracked,
// no VGPR round trip); fall back to explicit load+store staging.
__device__ __forceinline__ void async_copy16(const void* gsrc, void* ldst) {
#if defined(__gfx1250__) && __has_builtin(__builtin_amdgcn_global_load_async_to_lds_b128)
  __builtin_amdgcn_global_load_async_to_lds_b128(
      (__attribute__((address_space(1))) v4i_t*)gsrc,
      (__attribute__((address_space(3))) v4i_t*)ldst, 0, 0);
#else
  *(uint4*)ldst = *(const uint4*)gsrc;
#endif
}

__device__ __forceinline__ void wait_async_all() {
#if defined(__gfx1250__) && __has_builtin(__builtin_amdgcn_s_wait_asynccnt)
  __builtin_amdgcn_s_wait_asynccnt(0);
#elif defined(__gfx1250__)
  asm volatile("s_wait_asynccnt 0" ::: "memory");
#endif
}

// ---------------------------------------------------------------------------
// Kernel 1: convert W (blocks 0..1023) and X (blocks 1024..9215) rows to bf16,
// and compute exact f32 sum-of-squares per row.
// ---------------------------------------------------------------------------
__global__ __launch_bounds__(256) void prep_convert(
    const float* __restrict__ x, const float* __restrict__ w,
    unsigned short* __restrict__ xbf, unsigned short* __restrict__ wbf,
    float* __restrict__ xsq, float* __restrict__ wsq) {
  __shared__ float red[256];
  const int blk = blockIdx.x;
  const float* src; unsigned short* dst; float* sq;
  if (blk < K_CL) {
    src = w + (size_t)blk * D_DIM;
    dst = wbf + (size_t)blk * D_DIM;
    sq  = wsq + blk;
  } else {
    int row = blk - K_CL;
    src = x + (size_t)row * D_DIM;
    dst = xbf + (size_t)row * D_DIM;
    sq  = xsq + row;
  }
  const int t = threadIdx.x;
  float s = 0.0f;
  for (int e = t; e < D_DIM; e += 256) {
    float v = src[e];
    s += v * v;
    dst[e] = f32_to_bf16_rne(v);
  }
  red[t] = s;
  __syncthreads();
  for (int off = 128; off > 0; off >>= 1) {
    if (t < off) red[t] += red[t + off];
    __syncthreads();
  }
  if (t == 0) *sq = red[0];
}

// ---------------------------------------------------------------------------
// Kernel 2: S = X * W^T via bf16 WMMA (f32 accumulate), fused epilogue
// q_unnorm = 1 / (1 + max(0, xsq + wsq - 2S)).
// Block tile 128x128, 8 wave32 (4 wave-rows x 2 wave-cols), each wave owns
// 2 M-tiles x 4 N-tiles. Both X and W tiles double-buffered in LDS via
// async global->LDS copies.
// ---------------------------------------------------------------------------
__global__ __launch_bounds__(256) void gemm_epilogue(
    const unsigned short* __restrict__ xbf,
    const unsigned short* __restrict__ wbf,
    const float* __restrict__ xsq,
    const float* __restrict__ wsq,
    float* __restrict__ out) {
  __shared__ __align__(16) unsigned char ldsA[2][BM * TSTRIDE];
  __shared__ __align__(16) unsigned char ldsB[2][BN * TSTRIDE];

  const int t    = threadIdx.x;
  const int lane = t & 31;
  const int wave = t >> 5;
  const int wr   = wave & 3;   // 0..3 -> 32 rows each
  const int wc   = wave >> 2;  // 0..1 -> 64 cols each
  const int lm   = lane & 15;
  const int lh   = lane >> 4;

  const int mBase = blockIdx.x * BM;
  const int nBase = blockIdx.y * BN;

  // Per-thread staging assignment: 512 16B segments per tile, 2 per thread.
  const int r0 = t >> 1;                 // row 0..127
  const int s0 = (t & 1) * 2;           // segments {0,1} or {2,3}
  const unsigned short* gA0 = xbf + (size_t)(mBase + r0) * D_DIM + s0 * 8;
  const unsigned short* gB0 = wbf + (size_t)(nBase + r0) * D_DIM + s0 * 8;
  const int l0 = r0 * TSTRIDE + s0 * 16;

  v8f acc[2][4];
  #pragma unroll
  for (int i = 0; i < 2; i++)
    #pragma unroll
    for (int j = 0; j < 4; j++)
      #pragma unroll
      for (int e = 0; e < 8; e++)
        acc[i][j][e] = 0.0f;

  // Prologue: stage chunk 0 into buffer 0.
  async_copy16(gA0,      &ldsA[0][l0]);
  async_copy16(gA0 + 8,  &ldsA[0][l0 + 16]);
  async_copy16(gB0,      &ldsB[0][l0]);
  async_copy16(gB0 + 8,  &ldsB[0][l0 + 16]);

  for (int ch = 0; ch < NCHUNK; ch++) {
    const int buf = ch & 1;
    wait_async_all();       // my copies for chunk `ch` (and any later) done
    __syncthreads();        // everyone's copies done; previous buffer fully read

    if (ch + 1 < NCHUNK) {  // prefetch next chunk into the other buffer
      const int ko = (ch + 1) * BK;
      async_copy16(gA0 + ko,      &ldsA[buf ^ 1][l0]);
      async_copy16(gA0 + ko + 8,  &ldsA[buf ^ 1][l0 + 16]);
      async_copy16(gB0 + ko,      &ldsB[buf ^ 1][l0]);
      async_copy16(gB0 + ko + 8,  &ldsB[buf ^ 1][l0 + 16]);
    }

    // A fragments per ISA 16-bit A 16x32 layout:
    // v0..3 = K[lh*8 .. +7], v4..7 = K[16 + lh*8 .. +7]
    union { uint4 u[2]; v16bf v; } afr[2];
    #pragma unroll
    for (int mt = 0; mt < 2; mt++) {
      const unsigned char* ap = &ldsA[buf][(wr * 32 + mt * 16 + lm) * TSTRIDE + lh * 16];
      afr[mt].u[0] = *(const uint4*)(ap);
      afr[mt].u[1] = *(const uint4*)(ap + 32);
    }

    #pragma unroll
    for (int nt = 0; nt < 4; nt++) {
      // B fragment: lane holds column n = lm, K = lh*8 + 0..7 and 16 + lh*8 + 0..7.
      union { uint4 u[2]; v16bf v; } bfr;
      const unsigned char* bp = &ldsB[buf][(wc * 64 + nt * 16 + lm) * TSTRIDE + lh * 16];
      bfr.u[0] = *(const uint4*)(bp);
      bfr.u[1] = *(const uint4*)(bp + 32);
      #pragma unroll
      for (int mt = 0; mt < 2; mt++) {
        acc[mt][nt] = __builtin_amdgcn_wmma_f32_16x16x32_bf16(
            false, afr[mt].v, false, bfr.v, (short)0, acc[mt][nt], false, false);
      }
    }
  }

  // Epilogue: C/D layout -> lane: N = lm, VGPR r: M = lh*8 + r.
  #pragma unroll
  for (int mt = 0; mt < 2; mt++) {
    #pragma unroll
    for (int nt = 0; nt < 4; nt++) {
      int n = nBase + wc * 64 + nt * 16 + lm;
      float wv = wsq[n];
      #pragma unroll
      for (int r = 0; r < 8; r++) {
        int m = mBase + wr * 32 + mt * 16 + lh * 8 + r;
        float d = xsq[m] + wv - 2.0f * acc[mt][nt][r];
        d = fmaxf(d, 0.0f);
        out[(size_t)m * K_CL + n] = 1.0f / (1.0f + d);
      }
    }
  }
}

// ---------------------------------------------------------------------------
// Kernel 3: row-normalize q over the 1024 clusters, in place.
// ---------------------------------------------------------------------------
__global__ __launch_bounds__(256) void rownorm(float* __restrict__ out) {
  __shared__ float red[256];
  float* p = out + (size_t)blockIdx.x * K_CL;
  const int t = threadIdx.x;
  float v0 = p[t], v1 = p[t + 256], v2 = p[t + 512], v3 = p[t + 768];
  red[t] = v0 + v1 + v2 + v3;
  __syncthreads();
  for (int off = 128; off > 0; off >>= 1) {
    if (t < off) red[t] += red[t + off];
    __syncthreads();
  }
  float inv = 1.0f / red[0];
  p[t]       = v0 * inv;
  p[t + 256] = v1 * inv;
  p[t + 512] = v2 * inv;
  p[t + 768] = v3 * inv;
}

// ---------------------------------------------------------------------------
extern "C" void kernel_launch(void* const* d_in, const int* in_sizes, int n_in,
                              void* d_out, int out_size, void* d_ws, size_t ws_size,
                              hipStream_t stream) {
  const float* x = (const float*)d_in[0];   // [8192, 512]
  const float* w = (const float*)d_in[1];   // [1024, 512]
  float* out = (float*)d_out;               // [8192, 1024]

  unsigned char* ws = (unsigned char*)d_ws;
  unsigned short* wbf = (unsigned short*)(ws);
  unsigned short* xbf = (unsigned short*)(ws + (size_t)K_CL * D_DIM * 2);
  float* wsq = (float*)(ws + (size_t)K_CL * D_DIM * 2 + (size_t)B_ROWS * D_DIM * 2);
  float* xsq = wsq + K_CL;

  prep_convert<<<K_CL + B_ROWS, 256, 0, stream>>>(x, w, xbf, wbf, xsq, wsq);

  dim3 grid(B_ROWS / BM, K_CL / BN);
  gemm_epilogue<<<grid, 256, 0, stream>>>(xbf, wbf, xsq, wsq, out);

  rownorm<<<B_ROWS, 256, 0, stream>>>(out);
}
